// RecurrentClassifier_9328668967542
// MI455X (gfx1250) — matrix operand order repaired and checked by reading, not compiled, not run on hardware
//
#include <hip/hip_runtime.h>

// ---------------- problem constants ----------------
#define BB 32
#define TT 512
#define FF 1024
#define HH 1024
#define CC 2
#define MROWS (BB * TT)          // 16384
#define RNWG 16                  // workgroups in recurrence (grid barrier members)

typedef __attribute__((ext_vector_type(16))) __bf16 v16bf;
typedef __attribute__((ext_vector_type(8)))  float  v8f;

union BFrag {
    v16bf v;
    uint4 u[2];
};

__device__ __forceinline__ unsigned short f2bf(float f) {
    unsigned u = __float_as_uint(f);
    u += 0x7fffu + ((u >> 16) & 1u);      // round-to-nearest-even
    return (unsigned short)(u >> 16);
}

__device__ __forceinline__ v8f wmma_bf16(const v16bf a, const v16bf b, v8f c) {
    return __builtin_amdgcn_wmma_f32_16x16x32_bf16(
        /*neg_a=*/false, a, /*neg_b=*/false, b,
        /*c_mod=*/(short)0, c, /*reuse_a=*/false, /*reuse_b=*/false);
}

// Grid-wide barrier: phase-counter scheme; counter is reset per launch via memset.
__device__ __forceinline__ void grid_barrier(unsigned* bar, unsigned phase, unsigned nwg) {
    __threadfence();            // release all prior global writes
    __syncthreads();
    if (threadIdx.x == 0) {
        atomicAdd(bar, 1u);
        const unsigned target = phase * nwg;
        while (__hip_atomic_load(bar, __ATOMIC_RELAXED, __HIP_MEMORY_SCOPE_AGENT) < target) {
            __builtin_amdgcn_s_sleep(1);
        }
    }
    __syncthreads();
    __threadfence();            // acquire
}

// ---------------- f32 -> bf16 conversion ----------------
__global__ __launch_bounds__(256) void cvt_f32_bf16(const float* __restrict__ src,
                                                    unsigned short* __restrict__ dst, int n) {
    for (int i = blockIdx.x * blockDim.x + threadIdx.x; i < n; i += gridDim.x * blockDim.x)
        dst[i] = f2bf(src[i]);
}

// ---------------- projection GEMM ----------------
// out[M,N] = A[M,K](bf16) @ W[N,K](bf16)^T + bias_a[N] + bias_b[N]   (f32 accumulate)
// Block tile 128x128, 8 waves (4 in M x 2 in N), wave tile 32x64.
__global__ __launch_bounds__(256) void proj_gemm(const unsigned short* __restrict__ A,
                                                 const unsigned short* __restrict__ W,
                                                 const float* __restrict__ bias_a,
                                                 const float* __restrict__ bias_b,
                                                 float* __restrict__ out,
                                                 int M, int N, int K) {
    const int tid  = threadIdx.x;
    const int lane = tid & 31;
    const int wid  = tid >> 5;
    const int mw   = wid & 3;                 // wave row 0..3
    const int nw   = wid >> 2;                // wave col 0..1
    const int waveM = blockIdx.y * 128 + mw * 32;
    const int waveN = blockIdx.x * 128 + nw * 64;

    const int l15  = lane & 15;
    const int lhi  = lane >> 4;               // 0/1
    const int aoff = lhi * 16;                // A: lanes 0-15 -> K 0..15 ; 16-31 -> 16..31
    const int boff = lhi * 16;                // B: lanes 0-15 -> K 0..15 ; 16-31 -> 16..31

    const v8f vzero = {0.f, 0.f, 0.f, 0.f, 0.f, 0.f, 0.f, 0.f};
    v8f acc[2][4];
#pragma unroll
    for (int i = 0; i < 2; ++i)
#pragma unroll
        for (int j = 0; j < 4; ++j) acc[i][j] = vzero;

    for (int k0 = 0; k0 < K; k0 += 32) {
        BFrag a[2], b[4];
#pragma unroll
        for (int i = 0; i < 2; ++i) {
            const unsigned short* p = A + (size_t)(waveM + i * 16 + l15) * K + k0 + aoff;
            a[i].u[0] = *(const uint4*)p;
            a[i].u[1] = *(const uint4*)(p + 8);
        }
#pragma unroll
        for (int j = 0; j < 4; ++j) {
            const unsigned short* p = W + (size_t)(waveN + j * 16 + l15) * K + k0 + boff;
            b[j].u[0] = *(const uint4*)p;
            b[j].u[1] = *(const uint4*)(p + 8);
        }
#pragma unroll
        for (int i = 0; i < 2; ++i)
#pragma unroll
            for (int j = 0; j < 4; ++j) acc[i][j] = wmma_bf16(a[i].v, b[j].v, acc[i][j]);
    }

#pragma unroll
    for (int i = 0; i < 2; ++i)
#pragma unroll
        for (int j = 0; j < 4; ++j) {
            const int col = waveN + j * 16 + l15;
            const float bsum = bias_a[col] + bias_b[col];
#pragma unroll
            for (int r = 0; r < 8; ++r) {
                const int row = waveM + i * 16 + r + lhi * 8;   // C/D layout
                out[(size_t)row * N + col] = acc[i][j][r] + bsum;
            }
        }
}

// ---------------- persistent recurrence kernel ----------------
// h_t = tanh(pre[:,t,:] + h_{t-1} @ W_hh^T), 512 steps, 16 WGs, each owns 64 cols.
__global__ __launch_bounds__(256) void rnn_recur(const float* __restrict__ pre,        // [B,T,H]
                                                 const unsigned short* __restrict__ Wb, // W_hh bf16 [H,H]
                                                 const float* __restrict__ h0,          // [B,H]
                                                 unsigned short* __restrict__ hbuf,     // [2][B*H] bf16
                                                 unsigned short* __restrict__ y_bf16,   // [B,T,H] or null
                                                 float* __restrict__ y_f32,             // [B,T,H] or null
                                                 float* __restrict__ hn_out,            // [B,H]
                                                 unsigned* __restrict__ bar) {
    const int tid  = threadIdx.x;
    const int lane = tid & 31;
    const int wid  = tid >> 5;
    const int mi   = wid & 1;                 // M tile 0..1 (batch 0-15 / 16-31)
    const int ni   = wid >> 1;                // N tile 0..3 within the 64-col slice
    const int ncol0 = blockIdx.x * 64;

    const int l15  = lane & 15;
    const int lhi  = lane >> 4;
    const int aoff = lhi * 16;
    const int boff = lhi * 16;
    const int ngl  = ncol0 + ni * 16 + l15;   // global output column for this lane

    __shared__ unsigned short hlds[BB * HH];  // 64 KB: h_{t-1} staged in bf16

    // init hbuf[0] slice from f32 h0
    for (int idx = tid; idx < BB * 64; idx += 256) {
        const int b = idx >> 6;
        const int c = idx & 63;
        hbuf[b * HH + ncol0 + c] = f2bf(h0[b * HH + ncol0 + c]);
    }
    unsigned phase = 0;
    grid_barrier(bar, ++phase, RNWG);

    const v8f vzero = {0.f, 0.f, 0.f, 0.f, 0.f, 0.f, 0.f, 0.f};

    for (int t = 0; t < TT; ++t) {
        const unsigned short* hcur  = hbuf + (size_t)(t & 1) * BB * HH;
        unsigned short*       hnext = hbuf + (size_t)((t + 1) & 1) * BB * HH;

        // stage h_{t-1} (64 KB) into LDS, fully coalesced b128s
        {
            const uint4* src = (const uint4*)hcur;
            uint4*       dst = (uint4*)hlds;
            for (int idx = tid; idx < (BB * HH) / 8; idx += 256) dst[idx] = src[idx];
        }
        __syncthreads();

        // accumulator init from pre (which already carries b_ih + b_hh)
        v8f acc = vzero;
#pragma unroll
        for (int r = 0; r < 8; ++r) {
            const int m = mi * 16 + r + lhi * 8;          // batch row, C/D layout
            acc[r] = pre[((size_t)m * TT + t) * HH + ngl];
        }

        // 16x16x1024 slice of the step GEMM: 32 bf16 WMMAs
        for (int k0 = 0; k0 < HH; k0 += 32) {
            BFrag a, b;
            const unsigned short* pa = hlds + (mi * 16 + l15) * HH + k0 + aoff;
            a.u[0] = *(const uint4*)pa;
            a.u[1] = *(const uint4*)(pa + 8);
            const unsigned short* pb = Wb + (size_t)ngl * HH + k0 + boff;
            b.u[0] = *(const uint4*)pb;
            b.u[1] = *(const uint4*)(pb + 8);
            acc = wmma_bf16(a.v, b.v, acc);
        }

        // tanh, publish h_t (bf16 double buffer) and layer output
#pragma unroll
        for (int r = 0; r < 8; ++r) {
            const int m = mi * 16 + r + lhi * 8;
            const float hv = tanhf(acc[r]);
            hnext[m * HH + ngl] = f2bf(hv);
            if (y_bf16) y_bf16[((size_t)m * TT + t) * HH + ngl] = f2bf(hv);
            else        y_f32 [((size_t)m * TT + t) * HH + ngl] = hv;
            if (t == TT - 1) hn_out[m * HH + ngl] = hv;
        }

        grid_barrier(bar, ++phase, RNWG);     // also protects LDS re-staging (WAR)
    }
}

// ---------------- classifier ----------------
__global__ __launch_bounds__(256) void classifier(const float* __restrict__ y1,
                                                  const float* __restrict__ Wc,
                                                  const float* __restrict__ bc,
                                                  float* __restrict__ out) {
    __shared__ float red[256];
    const int b = blockIdx.x >> 1;
    const int c = blockIdx.x & 1;
    const float* yr = y1 + (size_t)b * TT * HH;
    const float* wr = Wc + (size_t)c * TT * HH;
    float s = 0.f;
    for (int i = threadIdx.x; i < TT * HH; i += 256) s += yr[i] * wr[i];
    red[threadIdx.x] = s;
    __syncthreads();
    for (int off = 128; off > 0; off >>= 1) {
        if (threadIdx.x < off) red[threadIdx.x] += red[threadIdx.x + off];
        __syncthreads();
    }
    if (threadIdx.x == 0) out[b * CC + c] = red[0] + bc[c];
}

// ---------------- host side ----------------
extern "C" void kernel_launch(void* const* d_in, const int* in_sizes, int n_in,
                              void* d_out, int out_size, void* d_ws, size_t ws_size,
                              hipStream_t stream) {
    (void)in_sizes; (void)n_in; (void)out_size; (void)ws_size;

    const float* x     = (const float*)d_in[0];
    const float* h0    = (const float*)d_in[1];
    const float* W_ih0 = (const float*)d_in[2];
    const float* W_hh0 = (const float*)d_in[3];
    const float* b_ih0 = (const float*)d_in[4];
    const float* b_hh0 = (const float*)d_in[5];
    const float* W_ih1 = (const float*)d_in[6];
    const float* W_hh1 = (const float*)d_in[7];
    const float* b_ih1 = (const float*)d_in[8];
    const float* b_hh1 = (const float*)d_in[9];
    const float* W_cls = (const float*)d_in[10];
    const float* b_cls = (const float*)d_in[11];
    float* out = (float*)d_out;                        // [64 logits][2*32*1024 h_n]

    char* ws = (char*)d_ws;
    size_t o = 0;
    unsigned short* Wb_ih0 = (unsigned short*)(ws + o); o += (size_t)HH * FF * 2;
    unsigned short* Wb_hh0 = (unsigned short*)(ws + o); o += (size_t)HH * HH * 2;
    unsigned short* Wb_ih1 = (unsigned short*)(ws + o); o += (size_t)HH * HH * 2;
    unsigned short* Wb_hh1 = (unsigned short*)(ws + o); o += (size_t)HH * HH * 2;
    unsigned short* xb     = (unsigned short*)(ws + o); o += (size_t)MROWS * FF * 2;
    unsigned short* y0b    = (unsigned short*)(ws + o); o += (size_t)MROWS * HH * 2;
    float*          pre    = (float*)(ws + o);          o += (size_t)MROWS * HH * 4;  // reused by both layers
    float*          y1     = (float*)(ws + o);          o += (size_t)MROWS * HH * 4;
    unsigned short* hbuf   = (unsigned short*)(ws + o); o += (size_t)2 * BB * HH * 2;
    unsigned*       bars   = (unsigned*)(ws + o);       o += 256;

    hipMemsetAsync(bars, 0, 256, stream);

    // bf16 conversions
    cvt_f32_bf16<<<2048, 256, 0, stream>>>(x,     xb,     MROWS * FF);
    cvt_f32_bf16<<<512,  256, 0, stream>>>(W_ih0, Wb_ih0, HH * FF);
    cvt_f32_bf16<<<512,  256, 0, stream>>>(W_hh0, Wb_hh0, HH * HH);
    cvt_f32_bf16<<<512,  256, 0, stream>>>(W_ih1, Wb_ih1, HH * HH);
    cvt_f32_bf16<<<512,  256, 0, stream>>>(W_hh1, Wb_hh1, HH * HH);

    const dim3 pg(HH / 128, MROWS / 128);   // (8, 128)

    // layer 0
    proj_gemm<<<pg, 256, 0, stream>>>(xb, Wb_ih0, b_ih0, b_hh0, pre, MROWS, HH, FF);
    rnn_recur<<<RNWG, 256, 0, stream>>>(pre, Wb_hh0, h0, hbuf,
                                        y0b, nullptr, out + BB * CC, bars + 0);
    // layer 1
    proj_gemm<<<pg, 256, 0, stream>>>(y0b, Wb_ih1, b_ih1, b_hh1, pre, MROWS, HH, HH);
    rnn_recur<<<RNWG, 256, 0, stream>>>(pre, Wb_hh1, h0 + (size_t)BB * HH, hbuf,
                                        nullptr, y1, out + BB * CC + (size_t)BB * HH, bars + 16);
    // classifier
    classifier<<<BB * CC, 256, 0, stream>>>(y1, W_cls, b_cls, out);
}
